// SelfAttention_38302518345838
// MI455X (gfx1250) — matrix operand order repaired and checked by reading, not compiled, not run on hardware
//
#include <hip/hip_runtime.h>

#define BATCH  4
#define SLEN   2048
#define DMODEL 1024
#define QPITCH 1032   // bf16 row pitch: 516 dwords == 4 (mod 64)  -> conflict-free 16B reads
#define VPITCH 40     // bf16 row pitch: 20 dwords, 20n mod 64 distinct for n=0..15

typedef __attribute__((ext_vector_type(4)))  __bf16       v4bf;
typedef __attribute__((ext_vector_type(8)))  __bf16       v8bf;
typedef __attribute__((ext_vector_type(16))) __bf16       v16bf;
typedef __attribute__((ext_vector_type(8)))  float        v8f;
typedef __attribute__((ext_vector_type(4)))  unsigned int u32x4;
typedef __attribute__((ext_vector_type(8)))  int          i32x8;
typedef __attribute__((ext_vector_type(4)))  int          i32x4;

#define SHUF16(lo, hi) __builtin_shufflevector(lo, hi, 0,1,2,3,4,5,6,7,8,9,10,11,12,13,14,15)
#define WMMA_BF16(a, b, c) \
  __builtin_amdgcn_wmma_f32_16x16x32_bf16(false, a, false, b, (short)0, c, false, false)

__global__ __launch_bounds__(128, 1)
void SelfAttention_38302518345838_kernel(const float* __restrict__ x,
                                         const float* __restrict__ mask,
                                         float* __restrict__ out) {
  __shared__ __attribute__((aligned(16))) __bf16 Qs[4][16][QPITCH]; // 132096 B
  __shared__ __attribute__((aligned(16))) __bf16 Ks[32][QPITCH];    //  66048 B
  __shared__ __attribute__((aligned(16))) __bf16 Vt[DMODEL][VPITCH];//  81920 B
  __shared__ __attribute__((aligned(16))) float  Ps[4][16][32];     //   8192 B
  __shared__ __attribute__((aligned(16))) float  Ms[SLEN];          //   8192 B

  const int wid  = threadIdx.x >> 5;
  const int lane = threadIdx.x & 31;
  const int m    = lane & 15;   // A row / B col / C col
  const int hf   = lane >> 4;   // lane half
  const int ak   = hf * 8;      // A-operand K base
  const int bk   = hf * 8;      // B-operand K base

  const int b     = blockIdx.x >> 5;         // 32 query blocks per batch
  const int qbase = (blockIdx.x & 31) * 64;  // 64 q rows per WG (16 per wave)

  const float* Xb = x + (size_t)b * SLEN * DMODEL;

  // ---- mask row for this batch -> LDS via Tensor Data Mover (TDM) ----
#if __has_builtin(__builtin_amdgcn_tensor_load_to_lds)
  if (threadIdx.x == 0) {
    unsigned long long ga = (unsigned long long)(size_t)(mask + (size_t)b * SLEN);
    unsigned ldsoff = (unsigned)(size_t)(&Ms[0]);  // low 32 bits of flat LDS ptr = LDS offset
    u32x4 g0 = { 1u,                                    // count = 1 valid descriptor
                 ldsoff,                                // lds_addr
                 (unsigned)(ga & 0xffffffffu),          // global_addr[31:0]
                 (unsigned)((ga >> 32) & 0x01ffffffu) | 0x80000000u }; // addr[56:32] | type=2
    i32x8 g1 = { (2 << 16),        // data_size = 4 bytes
                 (SLEN << 16),     // tensor_dim0 = 2048  (bits 79:48)
                 (1 << 16),        // tensor_dim1 = 1     (bits 111:80)
                 (SLEN << 16),     // tile_dim0   = 2048  (bits 127:112)
                 1,                // tile_dim1 = 1, tile_dim2 = 0
                 SLEN,             // tensor_dim0_stride
                 0, 0 };
    i32x4 g2 = { 1, 0, 0, 0 };     // tensor_dim2 = 1
    i32x4 g3 = { 0, 0, 0, 0 };
    i32x8 g4 = { 0, 0, 0, 0, 0, 0, 0, 0 };  // extra group on clang-23 builtin (zero-filled)
    __builtin_amdgcn_tensor_load_to_lds(g0, g1, g2, g3, g4, 0);
    __builtin_amdgcn_s_wait_tensorcnt(0);
  }
#else
  for (int i = threadIdx.x; i < SLEN; i += 128) Ms[i] = mask[(size_t)b * SLEN + i];
#endif

  // ---- stage this wave's 16 Q rows to LDS as bf16 (row-major, padded) ----
  for (int rr = 0; rr < 16; ++rr) {
    const float* src = Xb + (size_t)(qbase + wid * 16 + rr) * DMODEL;
    for (int c = lane * 4; c < DMODEL; c += 128) {
      float4 v = *(const float4*)(src + c);
      v4bf y = { (__bf16)v.x, (__bf16)v.y, (__bf16)v.z, (__bf16)v.w };
      *(v4bf*)&Qs[wid][rr][c] = y;
    }
  }

  // flash-attention state: O accumulator (16x1024 f32), per-row max / sum
  v8f o[64];
  float mrow[8], lrow[8];
#pragma unroll
  for (int nt = 0; nt < 64; ++nt) { v8f z = {}; o[nt] = z; }
#pragma unroll
  for (int i = 0; i < 8; ++i) { mrow[i] = -3.0e38f; lrow[i] = 0.0f; }

  const float L2E = 1.44269504088896341f;
  const float cs  = 0.03125f * L2E;  // (1/sqrt(1024)) * log2(e)

  for (int kb = 0; kb < SLEN; kb += 32) {
    __syncthreads();  // previous tile fully consumed before restage
    // ---- cooperatively stage 32 K/V rows: row-major Ks + transposed Vt ----
    for (int rr = 0; rr < 8; ++rr) {
      int r = wid * 8 + rr;
      const float* src = Xb + (size_t)(kb + r) * DMODEL;
      for (int c = lane * 4; c < DMODEL; c += 128) {
        float4 v = *(const float4*)(src + c);
        __bf16 y0 = (__bf16)v.x, y1 = (__bf16)v.y, y2 = (__bf16)v.z, y3 = (__bf16)v.w;
        v4bf y = { y0, y1, y2, y3 };
        *(v4bf*)&Ks[r][c] = y;
        Vt[c + 0][r] = y0; Vt[c + 1][r] = y1; Vt[c + 2][r] = y2; Vt[c + 3][r] = y3;
      }
    }
    __syncthreads();

    // ---- S = Q K^T : two 16x16 score tiles, K-dim = D in chunks of 32 ----
    v8f s0 = {}, s1 = {};
#pragma unroll 4
    for (int d = 0; d < DMODEL; d += 32) {
      v8bf alo = *(const v8bf*)&Qs[wid][m][d + ak];
      v8bf ahi = *(const v8bf*)&Qs[wid][m][d + ak + 16];
      v16bf a  = SHUF16(alo, ahi);
      v8bf b0l = *(const v8bf*)&Ks[m][d + bk];
      v8bf b0h = *(const v8bf*)&Ks[m][d + bk + 16];
      s0 = WMMA_BF16(a, SHUF16(b0l, b0h), s0);
      v8bf b1l = *(const v8bf*)&Ks[16 + m][d + bk];
      v8bf b1h = *(const v8bf*)&Ks[16 + m][d + bk + 16];
      s1 = WMMA_BF16(a, SHUF16(b1l, b1h), s1);
    }

    // ---- online softmax in exp2 domain ----
    float mm0 = Ms[kb + m] * L2E;
    float mm1 = Ms[kb + 16 + m] * L2E;
    float alpha[8];
#pragma unroll
    for (int i = 0; i < 8; ++i) {
      float z0 = s0[i] * cs + mm0;
      float z1 = s1[i] * cs + mm1;
      float t = fmaxf(z0, z1);
      t = fmaxf(t, __shfl_xor(t, 1));
      t = fmaxf(t, __shfl_xor(t, 2));
      t = fmaxf(t, __shfl_xor(t, 4));
      t = fmaxf(t, __shfl_xor(t, 8));
      float mn = fmaxf(mrow[i], t);
      alpha[i] = exp2f(mrow[i] - mn);
      mrow[i]  = mn;
      z0 = exp2f(z0 - mn);
      z1 = exp2f(z1 - mn);
      float rs = z0 + z1;
      rs += __shfl_xor(rs, 1);
      rs += __shfl_xor(rs, 2);
      rs += __shfl_xor(rs, 4);
      rs += __shfl_xor(rs, 8);
      lrow[i] = lrow[i] * alpha[i] + rs;
      Ps[wid][i + 8 * hf][m]      = z0;   // C layout -> LDS
      Ps[wid][i + 8 * hf][16 + m] = z1;
    }

    // ---- reload P in A layout (16x32), convert to bf16 ----
    float4 p0 = *(const float4*)&Ps[wid][m][ak];
    float4 p1 = *(const float4*)&Ps[wid][m][ak + 4];
    float4 p2 = *(const float4*)&Ps[wid][m][ak + 16];
    float4 p3 = *(const float4*)&Ps[wid][m][ak + 20];
    v16bf pa = { (__bf16)p0.x, (__bf16)p0.y, (__bf16)p0.z, (__bf16)p0.w,
                 (__bf16)p1.x, (__bf16)p1.y, (__bf16)p1.z, (__bf16)p1.w,
                 (__bf16)p2.x, (__bf16)p2.y, (__bf16)p2.z, (__bf16)p2.w,
                 (__bf16)p3.x, (__bf16)p3.y, (__bf16)p3.z, (__bf16)p3.w };

    // ---- O = diag(alpha)·O + P·V ----
#pragma unroll
    for (int nt = 0; nt < 64; ++nt) {
#pragma unroll
      for (int i = 0; i < 8; ++i) o[nt][i] *= alpha[i];
    }
#pragma unroll
    for (int nt = 0; nt < 64; ++nt) {
      v8bf vl = *(const v8bf*)&Vt[nt * 16 + m][bk];
      v8bf vh = *(const v8bf*)&Vt[nt * 16 + m][bk + 16];
      o[nt] = WMMA_BF16(pa, SHUF16(vl, vh), o[nt]);
    }
  }

  // ---- epilogue: O / l ----
  float inv[8];
#pragma unroll
  for (int i = 0; i < 8; ++i) inv[i] = 1.0f / lrow[i];
  float* ob = out + ((size_t)b * SLEN + qbase + wid * 16) * DMODEL;
#pragma unroll
  for (int nt = 0; nt < 64; ++nt) {
#pragma unroll
    for (int i = 0; i < 8; ++i) {
      ob[(size_t)(i + 8 * hf) * DMODEL + nt * 16 + m] = o[nt][i] * inv[i];
    }
  }
}

extern "C" void kernel_launch(void* const* d_in, const int* in_sizes, int n_in,
                              void* d_out, int out_size, void* d_ws, size_t ws_size,
                              hipStream_t stream) {
  (void)in_sizes; (void)n_in; (void)d_ws; (void)ws_size; (void)out_size;
  const float* x    = (const float*)d_in[0];
  const float* mask = (const float*)d_in[1];
  float* out        = (float*)d_out;
  dim3 grid(BATCH * (SLEN / 64));  // 128 WGs: 4 waves x 16 q-rows each
  dim3 block(128);
  SelfAttention_38302518345838_kernel<<<grid, block, 0, stream>>>(x, mask, out);
}
